// GroupedQueryAttention_49005576847964
// MI455X (gfx1250) — compile-verified, matrix operand order checked by reading, not run on hardware
//
#include <hip/hip_runtime.h>

// ---------------- problem constants (from reference) ----------------
#define B_   2
#define T_   2048
#define C_   2048
#define H_   32
#define G_   8
#define D_   64
#define KV_  (G_ * D_)        // 512
#define REP_ (H_ / G_)        // 4

// ---------------- vector types ----------------
typedef _Float16 h8_t  __attribute__((ext_vector_type(8)));
typedef _Float16 v16h  __attribute__((ext_vector_type(16)));
typedef float    f8_t  __attribute__((ext_vector_type(8)));

// =====================================================================
// Generic tiled GEMM:  C[M,N] = A[M,K] * B[K,N]   (f32 in, f16 WMMA, f32 out)
// BM=128, BN=64, BK=32; 256 threads = 8 waves in 4x2; each wave 32x32 (2x2 frags)
// N is a template parameter so epilogue/row strides fold into store offsets.
// =====================================================================
#define BM 128
#define BN 64
#define BK 32

template <int N>
__global__ __launch_bounds__(256) void gemm_wmma_kernel(
    const float* __restrict__ A, const float* __restrict__ Bm,
    float* __restrict__ Cm, int K)
{
    __shared__ _Float16 As[BM][BK];   // 8 KB, row-major [m][k]
    __shared__ _Float16 Bs[BN][BK];   // 4 KB, [n][k] (transposed on global read)

    const int tid  = threadIdx.x;
    const int lane = tid & 31;
    const int wave = tid >> 5;        // 0..7
    const int wm   = wave >> 1;       // 0..3 (M direction)
    const int wn   = wave & 1;        // 0..1 (N direction)
    const int lm   = lane & 15;
    const int lh   = lane >> 4;

    const int m0 = blockIdx.y * BM;
    const int n0 = blockIdx.x * BN;

    // A-tile read pattern: thread chunk -> (row, 8 consecutive k)
    const int ar0 = tid >> 2;             // 0..63   (chunk i=0)
    const int ac0 = (tid & 3) * 8;        // 0,8,16,24
    // B-tile read pattern: thread -> fixed column n, 8 k-values
    const int bn = tid & 63;              // 0..63
    const int bk = (tid >> 6) * 8;        // 0,8,16,24

    f8_t acc[2][2] = {};

    for (int k0 = 0; k0 < K; k0 += BK) {
        // prefetch next tiles into cache while this tile computes
        if (k0 + BK < K) {
            __builtin_prefetch(A + (size_t)(m0 + ar0) * K + k0 + BK + ac0, 0, 1);
            __builtin_prefetch(Bm + (size_t)(k0 + BK + bk) * N + (n0 + bn), 0, 1);
        }
        // ---- stage A tile (128x32): 2 chunks of 8 halves -> ds_store_b128 ----
        #pragma unroll
        for (int i = 0; i < 2; ++i) {
            int r  = ar0 + i * 64;
            const float* ap = A + (size_t)(m0 + r) * K + k0 + ac0;
            float4 a0 = *(const float4*)(ap);
            float4 a1 = *(const float4*)(ap + 4);
            h8_t hv = { (_Float16)a0.x, (_Float16)a0.y, (_Float16)a0.z, (_Float16)a0.w,
                        (_Float16)a1.x, (_Float16)a1.y, (_Float16)a1.z, (_Float16)a1.w };
            *(h8_t*)&As[r][ac0] = hv;
        }
        // ---- stage B tile (32x64) into Bs[n][k]: strided global reads,
        //      contiguous ds_store_b128 ----
        {
            const float* bp = Bm + (size_t)(k0 + bk) * N + (n0 + bn);
            float t0 = bp[0 * (size_t)N];
            float t1 = bp[1 * (size_t)N];
            float t2 = bp[2 * (size_t)N];
            float t3 = bp[3 * (size_t)N];
            float t4 = bp[4 * (size_t)N];
            float t5 = bp[5 * (size_t)N];
            float t6 = bp[6 * (size_t)N];
            float t7 = bp[7 * (size_t)N];
            h8_t hv = { (_Float16)t0, (_Float16)t1, (_Float16)t2, (_Float16)t3,
                        (_Float16)t4, (_Float16)t5, (_Float16)t6, (_Float16)t7 };
            *(h8_t*)&Bs[bn][bk] = hv;
        }
        __syncthreads();

        // ---- fragments + WMMA ----
        v16h af[2], bf[2];
        #pragma unroll
        for (int f = 0; f < 2; ++f) {
            int m = wm * 32 + f * 16 + lm;
            union { v16h v; h8_t h[2]; } ua;
            ua.h[0] = *(const h8_t*)&As[m][lh * 8];
            ua.h[1] = *(const h8_t*)&As[m][16 + lh * 8];
            af[f] = ua.v;
            int n = wn * 32 + f * 16 + lm;
            union { v16h v; h8_t h[2]; } ub;
            ub.h[0] = *(const h8_t*)&Bs[n][lh * 8];
            ub.h[1] = *(const h8_t*)&Bs[n][16 + lh * 8];
            bf[f] = ub.v;
        }
        #pragma unroll
        for (int fm = 0; fm < 2; ++fm)
            #pragma unroll
            for (int fn = 0; fn < 2; ++fn)
                acc[fm][fn] = __builtin_amdgcn_wmma_f32_16x16x32_f16(
                    false, af[fm], false, bf[fn], (short)0, acc[fm][fn], false, false);
        __syncthreads();
    }

    // ---- epilogue: base pointer + compile-time row offsets ----
    #pragma unroll
    for (int fm = 0; fm < 2; ++fm)
        #pragma unroll
        for (int fn = 0; fn < 2; ++fn) {
            float* cp = Cm + (size_t)(m0 + wm * 32 + fm * 16 + 8 * lh) * N
                           + (n0 + wn * 32 + fn * 16 + lm);
            #pragma unroll
            for (int r = 0; r < 8; ++r)
                cp[r * (size_t)N] = acc[fm][fn][r];
        }
}

// =====================================================================
// RoPE on Q/K + f16 staging + f32 kv caches.  One block per (b,t) row.
// Vectorized: one thread processes 8 contiguous floats (4 RoPE pairs,
// all within one head since D=64 and chunks are 8-aligned).
// =====================================================================
__global__ __launch_bounds__(256) void rope_cache_kernel(
    const float* __restrict__ qf, const float* __restrict__ kf,
    const float* __restrict__ vf,
    const float* __restrict__ cosT, const float* __restrict__ sinT,
    _Float16* __restrict__ qh, _Float16* __restrict__ kh,
    _Float16* __restrict__ vh,
    float* __restrict__ kcache, float* __restrict__ vcache)
{
    const int row = blockIdx.x;           // 0..B*T-1
    const int t   = row & (T_ - 1);
    const int tid = threadIdx.x;

    // ---- Q: 2048 floats = 256 chunks of 8; one chunk per thread ----
    {
        const int off = tid * 8;                  // 0..2040
        const int ih0 = (off & (D_ - 1)) >> 1;    // 0,4,...,28 (pair idx in head)
        const float* qp = qf + (size_t)row * C_ + off;
        float4 x01 = *(const float4*)(qp);
        float4 x23 = *(const float4*)(qp + 4);
        float4 cv  = *(const float4*)(cosT + t * 32 + ih0);
        float4 sv  = *(const float4*)(sinT + t * 32 + ih0);
        h8_t hv = {
            (_Float16)(x01.x * cv.x - x01.y * sv.x),
            (_Float16)(x01.x * sv.x + x01.y * cv.x),
            (_Float16)(x01.z * cv.y - x01.w * sv.y),
            (_Float16)(x01.z * sv.y + x01.w * cv.y),
            (_Float16)(x23.x * cv.z - x23.y * sv.z),
            (_Float16)(x23.x * sv.z + x23.y * cv.z),
            (_Float16)(x23.z * cv.w - x23.w * sv.w),
            (_Float16)(x23.z * sv.w + x23.w * cv.w) };
        *(h8_t*)(qh + (size_t)row * C_ + off) = hv;
    }
    // ---- K: 512 floats = 64 chunks; threads 0..63 (also f32 cache) ----
    if (tid < 64) {
        const int off = tid * 8;
        const int ih0 = (off & (D_ - 1)) >> 1;
        const float* kp = kf + (size_t)row * KV_ + off;
        float4 x01 = *(const float4*)(kp);
        float4 x23 = *(const float4*)(kp + 4);
        float4 cv  = *(const float4*)(cosT + t * 32 + ih0);
        float4 sv  = *(const float4*)(sinT + t * 32 + ih0);
        float4 r01 = { x01.x * cv.x - x01.y * sv.x,
                       x01.x * sv.x + x01.y * cv.x,
                       x01.z * cv.y - x01.w * sv.y,
                       x01.z * sv.y + x01.w * cv.y };
        float4 r23 = { x23.x * cv.z - x23.y * sv.z,
                       x23.x * sv.z + x23.y * cv.z,
                       x23.z * cv.w - x23.w * sv.w,
                       x23.z * sv.w + x23.w * cv.w };
        h8_t hv = { (_Float16)r01.x, (_Float16)r01.y, (_Float16)r01.z, (_Float16)r01.w,
                    (_Float16)r23.x, (_Float16)r23.y, (_Float16)r23.z, (_Float16)r23.w };
        *(h8_t*)(kh + (size_t)row * KV_ + off) = hv;
        float* kc = kcache + (size_t)row * KV_ + off;
        *(float4*)(kc)     = r01;
        *(float4*)(kc + 4) = r23;
    }
    // ---- V: 512 floats = 64 chunks; threads 64..127 ----
    else if (tid < 128) {
        const int off = (tid - 64) * 8;
        const float* vp = vf + (size_t)row * KV_ + off;
        float4 v01 = *(const float4*)(vp);
        float4 v23 = *(const float4*)(vp + 4);
        h8_t hv = { (_Float16)v01.x, (_Float16)v01.y, (_Float16)v01.z, (_Float16)v01.w,
                    (_Float16)v23.x, (_Float16)v23.y, (_Float16)v23.z, (_Float16)v23.w };
        *(h8_t*)(vh + (size_t)row * KV_ + off) = hv;
        float* vc = vcache + (size_t)row * KV_ + off;
        *(float4*)(vc)     = v01;
        *(float4*)(vc + 4) = v23;
    }
}

// =====================================================================
// Flash attention (causal, GQA). Block = 128 thr (4 waves, 2x2), tile 64x64.
// Q/K tiles use GLOBAL_LOAD_ASYNC_TO_LDS_B128 (ASYNCcnt) on gfx1250.
// K/V global pointers and LDS destinations are hoisted and bumped by a
// constant per kv-tile; prefetch of tile j+1 is issued under one branch.
// =====================================================================
#define BR 64
#define BCC 64
#define KV_TILE_STEP (BCC * KV_)   // 64 kv rows * 512 halves

__global__ __launch_bounds__(128) void attn_kernel(
    const _Float16* __restrict__ qh, const _Float16* __restrict__ kh,
    const _Float16* __restrict__ vh, float* __restrict__ of)
{
    __shared__ _Float16 Qs[BR][D_];     // 8 KB
    __shared__ _Float16 Ks[BCC][D_];    // 8 KB, [kv][d]
    __shared__ _Float16 VsT[D_][BCC];   // 8 KB, [d][kv] (transposed)
    __shared__ float    Ss[BR][BCC];    // 16 KB
    __shared__ _Float16 Ps[BR][BCC];    // 8 KB
    __shared__ float    mrow[BR], lrow[BR], frow[BR];

    const int qt = blockIdx.x, h = blockIdx.y, b = blockIdx.z;
    const int g  = h / REP_;
    const int q0 = qt * BR;

    const int tid  = threadIdx.x;
    const int lane = tid & 31;
    const int wave = tid >> 5;          // 0..3
    const int wm   = wave >> 1;         // 0..1 (q-row dir)
    const int wn   = wave & 1;          // 0..1 (col dir)
    const int lm   = lane & 15;
    const int lh   = lane >> 4;

    // ---- load Q tile: 64 x 64 halves, 16B chunks, 4 per thread ----
    #pragma unroll
    for (int u = 0; u < 4; ++u) {
        int i  = tid + u * 128;              // 0..511
        int r  = i >> 3;
        int c8 = (i & 7) * 8;
        const _Float16* gp = qh + (((size_t)(b * T_ + q0 + r)) * H_ + h) * D_ + c8;
#if defined(__gfx1250__)
        unsigned lds = (unsigned)(size_t)&Qs[r][c8];
        asm volatile("global_load_async_to_lds_b128 %0, %1, off"
                     :: "v"(lds), "v"(gp) : "memory");
#else
        *(h8_t*)&Qs[r][c8] = *(const h8_t*)gp;
#endif
    }
    if (tid < BR) { mrow[tid] = -3.0e38f; lrow[tid] = 0.0f; }

    // ---- hoisted per-thread K/V pointers + LDS destinations ----
    const _Float16* kp[4];
    const _Float16* vp[4];
    unsigned ldsK[4];
    #pragma unroll
    for (int u = 0; u < 4; ++u) {
        int i  = tid + u * 128;
        int r  = i >> 3;
        int c8 = (i & 7) * 8;
        size_t base = (((size_t)(b * T_ + r)) * G_ + g) * D_ + c8;   // j0 = 0
        kp[u]   = kh + base;
        vp[u]   = vh + base;
        ldsK[u] = (unsigned)(size_t)&Ks[r][c8];
    }
    const int vr  = tid >> 3;            // kv row handled for V-transpose, chunk 0
    const int vc8 = (tid & 7) * 8;

    f8_t Oa[2][2] = {};
#if defined(__gfx1250__)
    asm volatile("s_wait_asynccnt 0x0" ::: "memory");
#endif
    __syncthreads();

    for (int j = 0; j <= qt; ++j) {
        const int j0 = j * BCC;
        // ---- K tile: async copy global->LDS (straight-line, full EXEC) ----
        #pragma unroll
        for (int u = 0; u < 4; ++u) {
#if defined(__gfx1250__)
            asm volatile("global_load_async_to_lds_b128 %0, %1, off"
                         :: "v"(ldsK[u]), "v"(kp[u]) : "memory");
#else
            *(h8_t*)&Ks[(tid + u * 128) >> 3][((tid + u * 128) & 7) * 8] =
                *(const h8_t*)kp[u];
#endif
        }
        // ---- single-branch prefetch of next K/V tiles ----
        if (j < qt) {
            #pragma unroll
            for (int u = 0; u < 4; ++u) {
                __builtin_prefetch(kp[u] + KV_TILE_STEP, 0, 1);
                __builtin_prefetch(vp[u] + KV_TILE_STEP, 0, 1);
            }
        }
        // ---- V tile: manual transpose into VsT[d][kv] ----
        #pragma unroll
        for (int u = 0; u < 4; ++u) {
            h8_t vv8 = *(const h8_t*)vp[u];
            int r = vr + u * 16;
            #pragma unroll
            for (int uu = 0; uu < 8; ++uu) VsT[vc8 + uu][r] = vv8[uu];
        }
        // ---- advance hoisted pointers by one kv tile ----
        #pragma unroll
        for (int u = 0; u < 4; ++u) {
            kp[u] += KV_TILE_STEP;
            vp[u] += KV_TILE_STEP;
        }
#if defined(__gfx1250__)
        asm volatile("s_wait_asynccnt 0x0" ::: "memory");
#endif
        __syncthreads();

        // ---- S = Q * K^T (two K-steps of 32 over D=64) ----
        f8_t sc[2][2] = {};
        #pragma unroll
        for (int ks = 0; ks < 2; ++ks) {
            v16h aq[2], bk[2];
            #pragma unroll
            for (int f = 0; f < 2; ++f) {
                int m = wm * 32 + f * 16 + lm;
                union { v16h v; h8_t h[2]; } ua;
                ua.h[0] = *(const h8_t*)&Qs[m][ks * 32 + lh * 8];
                ua.h[1] = *(const h8_t*)&Qs[m][ks * 32 + 16 + lh * 8];
                aq[f] = ua.v;
                int n = wn * 32 + f * 16 + lm;
                union { v16h v; h8_t h[2]; } ub;
                ub.h[0] = *(const h8_t*)&Ks[n][ks * 32 + lh * 8];
                ub.h[1] = *(const h8_t*)&Ks[n][ks * 32 + 16 + lh * 8];
                bk[f] = ub.v;
            }
            #pragma unroll
            for (int fm = 0; fm < 2; ++fm)
                #pragma unroll
                for (int fn = 0; fn < 2; ++fn)
                    sc[fm][fn] = __builtin_amdgcn_wmma_f32_16x16x32_f16(
                        false, aq[fm], false, bk[fn], (short)0, sc[fm][fn], false, false);
        }

        // ---- write scaled + causal-masked scores to LDS ----
        const float scale = 0.125f;   // 1/sqrt(64)
        #pragma unroll
        for (int fm = 0; fm < 2; ++fm)
            #pragma unroll
            for (int fn = 0; fn < 2; ++fn)
                #pragma unroll
                for (int r = 0; r < 8; ++r) {
                    int ml = wm * 32 + fm * 16 + r + 8 * lh;
                    int nl = wn * 32 + fn * 16 + lm;
                    float sv = sc[fm][fn][r] * scale;
                    if (j0 + nl > q0 + ml) sv = -3.0e38f;
                    Ss[ml][nl] = sv;
                }
        __syncthreads();

        // ---- online softmax per row (threads 0..63) ----
        if (tid < BR) {
            int r = tid;
            float mprev = mrow[r];
            float mt = mprev;
            #pragma unroll 4
            for (int c2 = 0; c2 < BCC; ++c2) mt = fmaxf(mt, Ss[r][c2]);
            float sum = 0.0f;
            #pragma unroll 4
            for (int c2 = 0; c2 < BCC; ++c2) {
                float p = __expf(Ss[r][c2] - mt);
                Ps[r][c2] = (_Float16)p;
                sum += p;
            }
            float fac = __expf(mprev - mt);
            lrow[r] = lrow[r] * fac + sum;
            mrow[r] = mt;
            frow[r] = fac;
        }
        __syncthreads();

        // ---- rescale O accumulators ----
        #pragma unroll
        for (int fm = 0; fm < 2; ++fm)
            #pragma unroll
            for (int r = 0; r < 8; ++r) {
                int ml = wm * 32 + fm * 16 + r + 8 * lh;
                float fac = frow[ml];
                Oa[fm][0][r] *= fac;
                Oa[fm][1][r] *= fac;
            }

        // ---- O += P * V ----
        #pragma unroll
        for (int ks = 0; ks < 2; ++ks) {
            v16h ap[2], bv[2];
            #pragma unroll
            for (int f = 0; f < 2; ++f) {
                int m = wm * 32 + f * 16 + lm;
                union { v16h v; h8_t h[2]; } ua;
                ua.h[0] = *(const h8_t*)&Ps[m][ks * 32 + lh * 8];
                ua.h[1] = *(const h8_t*)&Ps[m][ks * 32 + 16 + lh * 8];
                ap[f] = ua.v;
                int n = wn * 32 + f * 16 + lm;
                union { v16h v; h8_t h[2]; } ub;
                ub.h[0] = *(const h8_t*)&VsT[n][ks * 32 + lh * 8];
                ub.h[1] = *(const h8_t*)&VsT[n][ks * 32 + 16 + lh * 8];
                bv[f] = ub.v;
            }
            #pragma unroll
            for (int fm = 0; fm < 2; ++fm)
                #pragma unroll
                for (int fn = 0; fn < 2; ++fn)
                    Oa[fm][fn] = __builtin_amdgcn_wmma_f32_16x16x32_f16(
                        false, ap[fm], false, bv[fn], (short)0, Oa[fm][fn], false, false);
        }
        __syncthreads();
    }

    // ---- epilogue: O /= l, write [B,T,H,D] as f32 (const row strides) ----
    #pragma unroll
    for (int fm = 0; fm < 2; ++fm)
        #pragma unroll
        for (int fn = 0; fn < 2; ++fn) {
            int ml0 = wm * 32 + fm * 16 + 8 * lh;
            int nl  = wn * 32 + fn * 16 + lm;
            float* op = of + (((size_t)(b * T_ + q0 + ml0)) * H_ + h) * D_ + nl;
            #pragma unroll
            for (int r = 0; r < 8; ++r) {
                float inv = 1.0f / lrow[ml0 + r];
                op[r * (size_t)(H_ * D_)] = Oa[fm][fn][r] * inv;
            }
        }
}

// =====================================================================
// Host-side launcher
// =====================================================================
extern "C" void kernel_launch(void* const* d_in, const int* in_sizes, int n_in,
                              void* d_out, int out_size, void* d_ws, size_t ws_size,
                              hipStream_t stream) {
    (void)in_sizes; (void)n_in; (void)out_size; (void)ws_size;

    const float* x  = (const float*)d_in[0];
    const float* fc = (const float*)d_in[1];
    const float* fs = (const float*)d_in[2];
    // d_in[3] = mask (non-None -> causal), not needed on device
    const float* Wq = (const float*)d_in[4];
    const float* Wk = (const float*)d_in[5];
    const float* Wv = (const float*)d_in[6];
    const float* Wo = (const float*)d_in[7];

    const size_t MT = (size_t)B_ * T_;   // 4096

    // workspace layout
    float* qf = (float*)d_ws;            // MT*C f32
    float* kf = qf + MT * C_;            // MT*KV f32
    float* vf = kf + MT * KV_;           // MT*KV f32
    float* of = vf + MT * KV_;           // MT*C  f32 (attention output)
    _Float16* qh = (_Float16*)(of + MT * C_);  // MT*C  f16
    _Float16* kh = qh + MT * C_;               // MT*KV f16
    _Float16* vh = kh + MT * KV_;              // MT*KV f16

    float* out    = (float*)d_out;       // [B,T,C]
    float* kcache = out + MT * C_;       // [B,T,G,D]
    float* vcache = kcache + MT * KV_;   // [B,T,G,D]

    dim3 blk256(256);

    // QKV projections
    gemm_wmma_kernel<C_><<<dim3(C_ / BN, MT / BM), blk256, 0, stream>>>(x, Wq, qf, C_);
    gemm_wmma_kernel<KV_><<<dim3(KV_ / BN, MT / BM), blk256, 0, stream>>>(x, Wk, kf, C_);
    gemm_wmma_kernel<KV_><<<dim3(KV_ / BN, MT / BM), blk256, 0, stream>>>(x, Wv, vf, C_);

    // RoPE + caches + f16 staging
    rope_cache_kernel<<<dim3((int)MT), blk256, 0, stream>>>(
        qf, kf, vf, fc, fs, qh, kh, vh, kcache, vcache);

    // causal GQA flash attention
    attn_kernel<<<dim3(T_ / BR, H_, B_), dim3(128), 0, stream>>>(qh, kh, vh, of);

    // output projection
    gemm_wmma_kernel<C_><<<dim3(C_ / BN, MT / BM), blk256, 0, stream>>>(of, Wo, out, C_);
}
